// aswjqUhyWZcYJGpzhqhoxBLqrSpNpcOM_84980222919379
// MI455X (gfx1250) — matrix operand order repaired and checked by reading, not compiled, not run on hardware
//
#include <hip/hip_runtime.h>
#include <hip/hip_bf16.h>

typedef _Float16 half_t;
typedef __attribute__((ext_vector_type(16))) _Float16 v16h;
typedef __attribute__((ext_vector_type(8)))  float    v8f;
typedef unsigned int u32x4 __attribute__((ext_vector_type(4)));
typedef int          i32x8 __attribute__((ext_vector_type(8)));
typedef int          i32x4 __attribute__((ext_vector_type(4)));

#define C_DIM  512
#define N_PIX  4096
#define BATCH  2
#define NGROUP 32
#define CPG    16
#define GEPS   1e-6f

#if __has_builtin(__builtin_amdgcn_tensor_load_to_lds) && \
    __has_builtin(__builtin_amdgcn_s_wait_tensorcnt)
#define HAVE_TDM 1
#else
#define HAVE_TDM 0
#endif

union AFrag { v16h v; half_t h[16]; };
union BFrag { v16h v; half_t h[16]; };
union CFrag { v8f  v; float  f[8];  };
union Pack8 { half_t h[8]; uint4 u; };
union U4    { uint4 u; unsigned short s[8]; };

// ---------------------------------------------------------------------------
// GroupNorm: x[B,C,N] f32 -> h16[B,C,N] f16
// ---------------------------------------------------------------------------
__global__ void __launch_bounds__(256)
gn_kernel(const float* __restrict__ x, const float* __restrict__ gw,
          const float* __restrict__ gb, half_t* __restrict__ h16) {
  __shared__ float red[256];
  __shared__ float red2[256];
  const int bg = blockIdx.x;
  const int b = bg / NGROUP, g = bg % NGROUP;
  const size_t base = ((size_t)b * C_DIM + (size_t)g * CPG) * N_PIX;
  const int tid = threadIdx.x;
  const int total = CPG * N_PIX;  // 65536
  float s = 0.f, s2 = 0.f;
  for (int e = tid; e < total; e += 256) {
    float v = x[base + e];
    s += v; s2 += v * v;
  }
  red[tid] = s; red2[tid] = s2;
  __syncthreads();
  for (int off = 128; off > 0; off >>= 1) {
    if (tid < off) { red[tid] += red[tid + off]; red2[tid] += red2[tid + off]; }
    __syncthreads();
  }
  const float inv = 1.0f / (float)total;
  const float mean = red[0] * inv;
  const float var  = red2[0] * inv - mean * mean;
  const float rstd = rsqrtf(var + GEPS);
  for (int e = tid; e < total; e += 256) {
    const int ch = g * CPG + (e >> 12);
    const float v = (x[base + e] - mean) * rstd * gw[ch] + gb[ch];
    h16[base + e] = (half_t)v;
  }
}

// ---------------------------------------------------------------------------
// GEMM: D[o,n] = sum_c W[o,c] * Bsrc[b,c,n] + bias[o]
// A tile row-major f16 (LDA=40), B tile COLUMN-major f16 (LDBK=40) so both
// WMMA fragment reads are contiguous 16B runs per lane -> ds_load_b128.
// mode 0: out16[b,o,n]            -> V
// mode 1: out16[b,n,o] (staged)   -> Q^T, K^T
// mode 2: out32[b,o,n] = resid+D  -> output projection
// ---------------------------------------------------------------------------
#define GM   64
#define GN   128
#define LDA  40
#define LDBK 40
#define SCR  24   // epilogue scratch stride (halves)

__global__ void __launch_bounds__(256)
gemm_kernel(const float* __restrict__ W, const float* __restrict__ bias,
            const half_t* __restrict__ Bsrc, half_t* __restrict__ out16,
            const float* __restrict__ resid, float* __restrict__ out32,
            int mode) {
  __shared__ half_t lds_a[GM * LDA];       // 5120 B
  __shared__ half_t lds_b[GN * LDBK];      // 10240 B (column-major tile)
  __shared__ half_t lds_s[8 * 64 * SCR];   // 24576 B epilogue scratch
  const int b  = blockIdx.z;
  const int n0 = blockIdx.x * GN;
  const int m0 = blockIdx.y * GM;
  const int tid = threadIdx.x;
  const int w = tid >> 5, lane = tid & 31;
  const int wm = w & 3, wn = w >> 2;
  const int lm = lane & 15, hi = lane >> 4;
  const size_t bB = (size_t)b * C_DIM * N_PIX;

  CFrag acc[4];
  #pragma unroll
  for (int s = 0; s < 4; ++s)
    #pragma unroll
    for (int r = 0; r < 8; ++r) acc[s].f[r] = 0.f;

  for (int k0 = 0; k0 < C_DIM; k0 += 32) {
    {  // A tile: 64 rows x 32 k, f32 -> f16, one b128 store per thread
      const int row = tid >> 2, kk = (tid & 3) * 8;
      const float4 f0 = *(const float4*)(W + (size_t)(m0 + row) * C_DIM + k0 + kk);
      const float4 f1 = *(const float4*)(W + (size_t)(m0 + row) * C_DIM + k0 + kk + 4);
      Pack8 p;
      p.h[0] = (half_t)f0.x; p.h[1] = (half_t)f0.y;
      p.h[2] = (half_t)f0.z; p.h[3] = (half_t)f0.w;
      p.h[4] = (half_t)f1.x; p.h[5] = (half_t)f1.y;
      p.h[6] = (half_t)f1.z; p.h[7] = (half_t)f1.w;
      *(uint4*)(lds_a + row * LDA + kk) = p.u;
    }
    {  // B tile: 32 k x 128 n -> column-major [n][k]; pack 2 k per b32 store
      const int kp = tid >> 4;
      const int nb = (tid & 15) * 8;
      U4 r0, r1;
      r0.u = *(const uint4*)(Bsrc + bB + (size_t)(k0 + 2 * kp) * N_PIX + n0 + nb);
      r1.u = *(const uint4*)(Bsrc + bB + (size_t)(k0 + 2 * kp + 1) * N_PIX + n0 + nb);
      unsigned int* dstw = (unsigned int*)lds_b;
      #pragma unroll
      for (int j = 0; j < 8; ++j)
        dstw[(nb + j) * (LDBK / 2) + kp] =
            (unsigned int)r0.s[j] | ((unsigned int)r1.s[j] << 16);
    }
    __syncthreads();

    AFrag a;
    #pragma unroll
    for (int i = 0; i < 16; ++i)
      a.h[i] = lds_a[(16 * wm + lm) * LDA + 8 * hi + (i & 7) + 16 * (i >> 3)];
    #pragma unroll
    for (int s = 0; s < 4; ++s) {
      BFrag bf;
      #pragma unroll
      for (int i = 0; i < 16; ++i)
        bf.h[i] = lds_b[(64 * wn + 16 * s + lm) * LDBK + 8 * hi + (i & 7) + 16 * (i >> 3)];
      acc[s].v = __builtin_amdgcn_wmma_f32_16x16x32_f16(
          false, a.v, false, bf.v, (short)0, acc[s].v, false, false);
    }
    __syncthreads();
  }

  if (mode == 1) {
    half_t* scr = lds_s + w * (64 * SCR);
    #pragma unroll
    for (int s = 0; s < 4; ++s)
      #pragma unroll
      for (int r = 0; r < 8; ++r) {
        const int o = m0 + 16 * wm + r + 8 * hi;
        scr[(s * 16 + lm) * SCR + r + 8 * hi] = (half_t)(acc[s].f[r] + bias[o]);
      }
    #pragma unroll
    for (int chunk = 0; chunk < 2; ++chunk) {
      const int nrow = chunk * 32 + lane;
      const uint4* src = (const uint4*)(scr + nrow * SCR);
      uint4* dst = (uint4*)(out16 + (size_t)b * N_PIX * C_DIM
                            + (size_t)(n0 + 64 * wn + nrow) * C_DIM + m0 + 16 * wm);
      dst[0] = src[0]; dst[1] = src[1];
    }
  } else {
    #pragma unroll
    for (int s = 0; s < 4; ++s)
      #pragma unroll
      for (int r = 0; r < 8; ++r) {
        const int o = m0 + 16 * wm + r + 8 * hi;
        const int n = n0 + 64 * wn + 16 * s + lm;
        const float v = acc[s].f[r] + bias[o];
        if (mode == 0)
          out16[bB + (size_t)o * N_PIX + n] = (half_t)v;
        else
          out32[bB + (size_t)o * N_PIX + n] = resid[bB + (size_t)o * N_PIX + n] + v;
      }
  }
}

// ---------------------------------------------------------------------------
// Flash attention. 128 thr (4 waves) per WG own 16 query rows.
// qT,kT are [N,C]; v is [C,N]; every fragment read is per-lane contiguous.
// K/V tiles are DOUBLE-BUFFERED via the Tensor Data Mover: wave0 issues the
// K tile, wave1 the V tile for iteration jb+1 while jb is being computed;
// s_wait_tensorcnt(1) retires the oldest in-flight DMA (in-order per wave).
// Uses 297 KB of the 320 KB WGP LDS. Fallback: coalesced uint4 copies.
// ---------------------------------------------------------------------------
#define QSH 520   // sQ stride (halves)
#define KSH 520   // sK stride (halves) = 512 + 8  (pad 4 dwords / row)
#define VSH 72    // sV stride (halves) = 64 + 8   (pad 4 dwords / row)
#define SSF 68    // sS stride (floats)
#define PSH 72    // sP stride (halves)

#define SKB 66560                    // 64*KSH*2
#define SVB 73728                    // 512*VSH*2
#define BUFB (SKB + SVB)             // 140288 per double-buffer stage
#define SK0_OFF 16640                // after 16*QSH*2 region
#define SV0_OFF (SK0_OFF + SKB)
#define SMEM_TOTAL (SK0_OFF + 2 * BUFB)  // 297216 B  (<320 KB WGP LDS)

__global__ void __launch_bounds__(128)
attn_kernel(const half_t* __restrict__ qT, const half_t* __restrict__ kT,
            const half_t* __restrict__ vM, half_t* __restrict__ oT) {
  extern __shared__ char smem[];
  half_t* sQ = (half_t*)smem;                   // 16 x QSH (overlaid later)
  float*  sS = (float*)smem;                    // 16 x SSF = 4352 B
  half_t* sP = (half_t*)(smem + 4352);          // 16 x PSH = 2304 B
  float*  st = (float*)(smem + 4352 + 2304);    // 48 floats
  float* mrow = st, * lrow = st + 16, * frow = st + 32;

  const int b  = blockIdx.y;
  const int i0 = blockIdx.x * 16;
  const int tid = threadIdx.x;
  const int w = tid >> 5, lane = tid & 31;
  const int lm = lane & 15, hi = lane >> 4;
  const float scale = 0.04419417382415922f;     // 512^-0.5
  const size_t bNC = (size_t)b * N_PIX * C_DIM;
  const size_t bCN = (size_t)b * C_DIM * N_PIX;

  {  // stage Q rows [16 x 512]
    const int row = tid >> 3, cc = (tid & 7) * 64;
    const uint4* src = (const uint4*)(qT + bNC + (size_t)(i0 + row) * C_DIM + cc);
    uint4* dst = (uint4*)(sQ + row * QSH + cc);
    #pragma unroll
    for (int q = 0; q < 8; ++q) dst[q] = src[q];
  }
  __syncthreads();

  AFrag qf[16];
  #pragma unroll
  for (int kc = 0; kc < 16; ++kc)
    #pragma unroll
    for (int i = 0; i < 16; ++i)
      qf[kc].h[i] = sQ[lm * QSH + kc * 32 + 8 * hi + (i & 7) + 16 * (i >> 3)];
  __syncthreads();  // sQ region may now be overlaid by sS/sP/st

  if (tid < 16) { mrow[tid] = -1e30f; lrow[tid] = 0.f; }

  CFrag acc[8];
  #pragma unroll
  for (int t = 0; t < 8; ++t)
    #pragma unroll
    for (int r = 0; r < 8; ++r) acc[t].f[r] = 0.f;

#if HAVE_TDM
  const unsigned ldsBase = (unsigned)__builtin_amdgcn_groupstaticsize();
  const i32x4 zg4 = {0, 0, 0, 0};
  const i32x8 zg8 = {0, 0, 0, 0, 0, 0, 0, 0};
  // K tile: tensor kT[4096,512] h16; tile 64 rows x 512; pad 4 dwords per
  //         256-dword row -> data_size=1(2B), pad_en, pad_interval=7, amt=3
  const i32x8 g1k = {(int)0x07D10000u,
                     (int)(512u << 16),        // tensor_dim0 = 512
                     (int)(4096u << 16),       // tensor_dim1 = 4096
                     (int)(512u << 16),        // tile_dim0 = 512
                     64,                       // tile_dim1 = 64
                     512, 0, 0};               // dim0_stride = 512
  // V tile: tensor v[512,4096] h16; tile 512 rows x 64; pad 4 dwords per
  //         32-dword row -> pad_interval=4, pad_amount=3
  const i32x8 g1v = {(int)0x07110000u,
                     (int)(4096u << 16),       // tensor_dim0 = 4096
                     (int)(512u << 16),        // tensor_dim1 = 512
                     (int)(64u << 16),         // tile_dim0 = 64
                     512,                      // tile_dim1 = 512
                     4096, 0, 0};              // dim0_stride = 4096

  auto issueK = [&](int j0, unsigned ldsOff) {
    const unsigned long long ga =
        (unsigned long long)(const void*)(kT + bNC + (size_t)j0 * C_DIM);
    const u32x4 g0 = {1u, ldsBase + ldsOff, (unsigned)ga,
                      (unsigned)(ga >> 32) | 0x80000000u};
    __builtin_amdgcn_tensor_load_to_lds(g0, g1k, zg4, zg4, zg8, 0);
  };
  auto issueV = [&](int j0, unsigned ldsOff) {
    const unsigned long long ga =
        (unsigned long long)(const void*)(vM + bCN + j0);
    const u32x4 g0 = {1u, ldsBase + ldsOff, (unsigned)ga,
                      (unsigned)(ga >> 32) | 0x80000000u};
    __builtin_amdgcn_tensor_load_to_lds(g0, g1v, zg4, zg4, zg8, 0);
  };

  // prologue: DMA tile 0 into buffer 0
  if (w == 0)      issueK(0, SK0_OFF);
  else if (w == 1) issueV(0, SV0_OFF);
#endif

  for (int jb = 0; jb < 64; ++jb) {
    const unsigned bufOff = (unsigned)(jb & 1) * BUFB;
    half_t* sK = (half_t*)(smem + SK0_OFF + bufOff);
    half_t* sV = (half_t*)(smem + SV0_OFF + bufOff);
#if HAVE_TDM
    if (jb + 1 < 64) {  // issue next tile into the other buffer, keep 1 in flight
      const int nj = (jb + 1) * 64;
      const unsigned nOff = (unsigned)((jb + 1) & 1) * BUFB;
      if (w == 0)      issueK(nj, SK0_OFF + nOff);
      else if (w == 1) issueV(nj, SV0_OFF + nOff);
      __builtin_amdgcn_s_wait_tensorcnt(1);   // oldest (tile jb) retired
    } else {
      __builtin_amdgcn_s_wait_tensorcnt(0);   // drain
    }
    __syncthreads();
#else
    {  // K^T tile: rows j0..j0+63 of kT, 64 x 512 halves
      const int j0 = jb * 64;
      const int row = tid >> 1, hh = (tid & 1) * 256;
      const uint4* src = (const uint4*)(kT + bNC + (size_t)(j0 + row) * C_DIM + hh);
      uint4* dst = (uint4*)(sK + row * KSH + hh);
      #pragma unroll
      for (int q = 0; q < 8; ++q) dst[q] = src[q];
      #pragma unroll
      for (int q = 0; q < 4; ++q) {          // V tile: 512 c rows x 64 j halves
        const int vrow = tid + 128 * q;
        const uint4* vs = (const uint4*)(vM + bCN + (size_t)vrow * N_PIX + j0);
        uint4* vd = (uint4*)(sV + vrow * VSH);
        #pragma unroll
        for (int u = 0; u < 8; ++u) vd[u] = vs[u];
      }
    }
    __syncthreads();
#endif

    // S tile (wave w -> columns [16w,16w+16)): B-frag = contiguous c run
    CFrag sc;
    #pragma unroll
    for (int r = 0; r < 8; ++r) sc.f[r] = 0.f;
    #pragma unroll
    for (int kc = 0; kc < 16; ++kc) {
      BFrag bf;
      #pragma unroll
      for (int i = 0; i < 16; ++i)
        bf.h[i] = sK[(w * 16 + lm) * KSH + kc * 32 + 8 * hi + (i & 7) + 16 * (i >> 3)];
      sc.v = __builtin_amdgcn_wmma_f32_16x16x32_f16(
          false, qf[kc].v, false, bf.v, (short)0, sc.v, false, false);
    }
    #pragma unroll
    for (int r = 0; r < 8; ++r)
      sS[(r + 8 * hi) * SSF + w * 16 + lm] = sc.f[r] * scale;
    __syncthreads();

    // online softmax (one thread per query row)
    if (tid < 16) {
      const int row = tid;
      const float mo = mrow[row];
      float mx = mo;
      for (int j = 0; j < 64; ++j) mx = fmaxf(mx, sS[row * SSF + j]);
      const float fac = __expf(mo - mx);
      float ln = lrow[row] * fac;
      for (int j = 0; j < 64; ++j) {
        const float p = __expf(sS[row * SSF + j] - mx);
        ln += p;
        sP[row * PSH + j] = (half_t)p;
      }
      mrow[row] = mx; lrow[row] = ln; frow[row] = fac;
    }
    __syncthreads();

    // rescale accumulator
    float fr[8];
    #pragma unroll
    for (int r = 0; r < 8; ++r) fr[r] = frow[r + 8 * hi];
    #pragma unroll
    for (int t = 0; t < 8; ++t)
      #pragma unroll
      for (int r = 0; r < 8; ++r) acc[t].f[r] *= fr[r];

    // PV: acc[16 x 128] += P[16x64] * V^T[64 x 128(w)]; B-frag = contiguous j
    #pragma unroll
    for (int ka = 0; ka < 2; ++ka) {
      AFrag pf;
      #pragma unroll
      for (int i = 0; i < 16; ++i)
        pf.h[i] = sP[lm * PSH + ka * 32 + 8 * hi + (i & 7) + 16 * (i >> 3)];
      #pragma unroll
      for (int t = 0; t < 8; ++t) {
        BFrag bf;
        #pragma unroll
        for (int i = 0; i < 16; ++i)
          bf.h[i] = sV[(w * 128 + t * 16 + lm) * VSH + ka * 32 + 8 * hi + (i & 7) + 16 * (i >> 3)];
        acc[t].v = __builtin_amdgcn_wmma_f32_16x16x32_f16(
            false, pf.v, false, bf.v, (short)0, acc[t].v, false, false);
      }
    }
    __syncthreads();
  }

  float li[8];
  #pragma unroll
  for (int r = 0; r < 8; ++r) li[r] = 1.0f / lrow[r + 8 * hi];

  // epilogue: stage [c][i] per wave in LDS (reuse K buffer 0), 32B stores
  half_t* scr = (half_t*)(smem + SK0_OFF) + w * (128 * SCR);
  #pragma unroll
  for (int t = 0; t < 8; ++t)
    #pragma unroll
    for (int r = 0; r < 8; ++r)
      scr[(t * 16 + lm) * SCR + r + 8 * hi] = (half_t)(acc[t].f[r] * li[r]);
  #pragma unroll
  for (int chunk = 0; chunk < 4; ++chunk) {
    const int crow = chunk * 32 + lane;
    const uint4* src = (const uint4*)(scr + crow * SCR);
    uint4* dst = (uint4*)(oT + bCN + (size_t)(w * 128 + crow) * N_PIX + i0);
    dst[0] = src[0]; dst[1] = src[1];
  }
}

// ---------------------------------------------------------------------------
extern "C" void kernel_launch(void* const* d_in, const int* in_sizes, int n_in,
                              void* d_out, int out_size, void* d_ws, size_t ws_size,
                              hipStream_t stream) {
  const float* x    = (const float*)d_in[0];
  const float* gn_w = (const float*)d_in[1];
  const float* gn_b = (const float*)d_in[2];
  const float* wq   = (const float*)d_in[3];
  const float* bq   = (const float*)d_in[4];
  const float* wk   = (const float*)d_in[5];
  const float* bk   = (const float*)d_in[6];
  const float* wv   = (const float*)d_in[7];
  const float* bv   = (const float*)d_in[8];
  const float* wo   = (const float*)d_in[9];
  const float* bo   = (const float*)d_in[10];
  float* out = (float*)d_out;

  char* ws = (char*)d_ws;
  const size_t S16 = (size_t)BATCH * C_DIM * N_PIX * sizeof(half_t);  // 8 MB
  half_t* h16 = (half_t*)(ws);
  half_t* qT  = (half_t*)(ws + 1 * S16);
  half_t* kT  = (half_t*)(ws + 2 * S16);
  half_t* vM  = (half_t*)(ws + 3 * S16);
  half_t* aT  = (half_t*)(ws + 4 * S16);

  gn_kernel<<<dim3(BATCH * NGROUP), 256, 0, stream>>>(x, gn_w, gn_b, h16);

  dim3 gg(N_PIX / GN, C_DIM / GM, BATCH);
  gemm_kernel<<<gg, 256, 0, stream>>>(wq, bq, h16, qT, nullptr, nullptr, 1);
  gemm_kernel<<<gg, 256, 0, stream>>>(wk, bk, h16, kT, nullptr, nullptr, 1);
  gemm_kernel<<<gg, 256, 0, stream>>>(wv, bv, h16, vM, nullptr, nullptr, 0);

  attn_kernel<<<dim3(N_PIX / 16, BATCH), 128, SMEM_TOTAL, stream>>>(qT, kT, vM, aT);

  gemm_kernel<<<gg, 256, 0, stream>>>(wo, bo, aT, nullptr, x, out, 2);
}
